// LSNNLayer_49117245997796
// MI455X (gfx1250) — matrix operand order repaired and checked by reading, not compiled, not run on hardware
//
#include <hip/hip_runtime.h>

// ---------------------------------------------------------------------------
// LSNN layer for MI455X (gfx1250, wave32, WMMA bf16 K=32).
// Persistent 16-WG kernel: each WG owns 64 hidden columns; weights + per-step
// A-frames live in 320KB LDS with bank-conflict-free padded rows (odd multiple
// of 16B). Frames staged each step via GLOBAL_LOAD_ASYNC_TO_LDS (ASYNCcnt).
// State lives in registers matching the WMMA C/D layout; per-step global
// arrive counters + 4-deep z ring buffer carry the recurrent dependency.
// ---------------------------------------------------------------------------

typedef __attribute__((ext_vector_type(16))) __bf16 v16bf;
typedef __attribute__((ext_vector_type(8)))  float  v8f;

#define T_STEPS 1000
#define BB 32
#define II 512
#define HH 1024
#define BH (BB*HH)                 // 32768
#define NWG 16
#define HWG 64                     // hidden columns per workgroup
#define LDS_ROWS (II+HH)           // 1536 K-rows of weights

// padded LDS row strides (elements) -> bank-conflict-free b128 access
#define WPAD 72                    // weight rows: 144 B (odd multiple of 16B)
#define ZPAD 1032                  // z frame rows: 2064 B
#define IPAD 520                   // input frame rows: 1040 B

#define ZST_OFF  (LDS_ROWS*WPAD*2)           // 221184
#define IST_OFF  (ZST_OFF + BB*ZPAD*2)       // 287232
#define LDS_BYTES (IST_OFF + BB*IPAD*2)      // 320512 <= 327680 (320KB)

// LSNN constants (norse defaults)
#define P_SYN  0.2f                 // dt * tau_syn_inv
#define P_MEM  0.1f                 // dt * tau_mem_inv
#define P_AD   (1.0e-3f/700.0f)     // dt * tau_adapt_inv
#define B_JUMP (1.8f/700.0f)        // beta * tau_adapt_inv

// workspace layout (bytes)
#define OFF_CNT 0                          // T counters (4 KB padded)
#define OFF_Z   4096                       // 4 slots x [B][H] bf16 = 256 KB
#define OFF_IWT (OFF_Z   + 4*BH*2)         // iw^T  [I][H] bf16 = 1 MB
#define OFF_RWT (OFF_IWT + II*HH*2)        // rw^T  [H][H] bf16 = 2 MB
#define OFF_INP (OFF_RWT + HH*HH*2)        // input [T][B][I] bf16 = 32 MB

#define WMMA_B16(ACC, A, B)                                                    \
  ACC = __builtin_amdgcn_wmma_f32_16x16x32_bf16(false, (A), false, (B),        \
                                                (short)0, (ACC), false, false)

// async global->LDS copy, 16 bytes per lane (GV mode, ASYNCcnt-tracked)
__device__ __forceinline__ void async_g2l_b128(unsigned lds_off,
                                               const void* gaddr) {
  asm volatile("global_load_async_to_lds_b128 %0, %1, off"
               :: "v"(lds_off), "v"(gaddr) : "memory");
}
__device__ __forceinline__ void wait_async0() {
  asm volatile("s_wait_asynccnt 0x0" ::: "memory");
}

// ---- prep kernels ----------------------------------------------------------

__global__ void k_init(unsigned* __restrict__ cnt, const float* __restrict__ z0,
                       __bf16* __restrict__ zsl) {
  int i = blockIdx.x * blockDim.x + threadIdx.x;
  if (i < T_STEPS) cnt[i] = 0u;                   // arrive counters
  if (i < BH)      zsl[3*BH + i] = (__bf16)z0[i]; // slot 3 holds z_{-1} = z0
}

__global__ void k_cvt(const float* __restrict__ src, __bf16* __restrict__ dst,
                      int n) {
  int i = blockIdx.x * blockDim.x + threadIdx.x;
  int stride = gridDim.x * blockDim.x;
  for (; i < n; i += stride) dst[i] = (__bf16)src[i];
}

__global__ void k_tw(const float* __restrict__ iw, const float* __restrict__ rw,
                     __bf16* __restrict__ iwT, __bf16* __restrict__ rwT) {
  const int NI = II*HH, NR = HH*HH;
  int i = blockIdx.x * blockDim.x + threadIdx.x;
  int stride = gridDim.x * blockDim.x;
  for (; i < NI + NR; i += stride) {
    if (i < NI) {
      int k = i / HH, h = i % HH;                 // iwT[k][h] = iw[h][k]
      iwT[i] = (__bf16)iw[(size_t)h*II + k];
    } else {
      int j = i - NI; int k = j / HH, h = j % HH;
      rwT[j] = (__bf16)rw[(size_t)h*HH + k];
    }
  }
}

// ---- persistent scan kernel ------------------------------------------------

__global__ __launch_bounds__(128, 1)
void lsnn_scan(const __bf16* __restrict__ inp_b,   // [T][B][I] bf16
               const __bf16* __restrict__ iwT_g,   // [I][H] bf16
               const __bf16* __restrict__ rwT_g,   // [H][H] bf16
               const float*  __restrict__ v0,
               const float*  __restrict__ i0,
               const float*  __restrict__ b0,
               __bf16*       __restrict__ zslots,  // [4][B][H] bf16 ring
               unsigned*     __restrict__ cnt,     // [T]
               float*        __restrict__ out)     // outputs + (zf,vf,if,bf)
{
  extern __shared__ __bf16 ldsw[];        // [LDS_ROWS][WPAD] weights, K-major
  __bf16* zst = ldsw + LDS_ROWS*WPAD;     // staged z frame   [32][ZPAD]
  __bf16* ist = zst + BB*ZPAD;            // staged input frm [32][IPAD]

  const int tid  = threadIdx.x;
  const int wg   = blockIdx.x;
  const int wave = tid >> 5;
  const int lane = tid & 31;
  const int l16  = lane >> 4;
  const int ln   = lane & 15;

  // ---- stage this WG's weight columns into padded LDS rows (one-time) ----
  {
    const int pairs = LDS_ROWS * (HWG/2);
    for (int p = tid; p < pairs; p += 128) {
      const int r = p >> 5;             // K row
      const int c = (p & 31) << 1;      // even column pair
      unsigned w;
      if (r < II) w = *(const unsigned*)(iwT_g + (size_t)r*HH + wg*HWG + c);
      else        w = *(const unsigned*)(rwT_g + (size_t)(r-II)*HH + wg*HWG + c);
      *(unsigned*)(&ldsw[r*WPAD + c]) = w;
    }
  }
  __syncthreads();

  const int hn = wg*HWG + wave*16 + ln;   // this lane's hidden column

  // state registers, element e = tile*8 + j maps to WMMA C/D layout:
  //   batch row b = tile*16 + j + 8*l16, column = hn
  float v_s[16], i_s[16], b_s[16], z_s[16];
#pragma unroll
  for (int e = 0; e < 16; ++e) {
    const int bi = (e >> 3)*16 + (e & 7) + 8*l16;
    const size_t a = (size_t)bi*HH + hn;
    v_s[e] = v0[a]; i_s[e] = i0[a]; b_s[e] = b0[a]; z_s[e] = 0.0f;
  }

  const int arow = ln;        // A-operand row (M) for tile 0
  const int kofs = l16 * 8;   // per-lane K offset within chunk

  for (int t = 0; t < T_STEPS; ++t) {
    // ---- LIF dynamics: spike, reset, adapt; acc <- i_dec ----
    v8f acc0, acc1;
#pragma unroll
    for (int e = 0; e < 16; ++e) {
      const float vd = v_s[e] + P_MEM * (i_s[e] - v_s[e]);
      const float bd = b_s[e] + P_AD  * (1.0f - b_s[e]);
      const float z  = (vd - bd) > 0.0f ? 1.0f : 0.0f;
      v_s[e] = (1.0f - z) * vd;
      b_s[e] = bd + z * B_JUMP;
      z_s[e] = z;
      const float idec = i_s[e] * (1.0f - P_SYN);
      if (e < 8) acc0[e] = idec; else acc1[e - 8] = idec;
      const int bi = (e >> 3)*16 + (e & 7) + 8*l16;
      const size_t a = (size_t)bi*HH + hn;
      out[(size_t)t*BH + a] = z;                          // spike output
      zslots[(size_t)(t & 3)*BH + a] = (__bf16)z;         // publish z_t
    }

    __threadfence();
    __syncthreads();   // all waves done with previous GEMM + this step's stores

    // ---- signal arrival ASAP so peers don't stall on us ----
    if (tid == 0)
      __hip_atomic_fetch_add(&cnt[t], 1u, __ATOMIC_RELEASE,
                             __HIP_MEMORY_SCOPE_AGENT);

    // ---- stage input[t] into LDS (independent of z: overlaps the spin) ----
    {
      const char* g = (const char*)(inp_b + (size_t)t*BB*II);   // 32768 B
#pragma unroll
      for (int j = 0; j < 16; ++j) {           // 2048 b128 chunks / 128 thr
        const int p   = j*128 + tid;
        const int row = p >> 6;                // 64 chunks per 1024B row
        const int col = (p & 63) << 4;
        async_g2l_b128(IST_OFF + (unsigned)(row*IPAD*2 + col),
                       g + (size_t)p*16u);
      }
    }

    if (t + 1 < T_STEPS)   // pull next input frame toward L2
      __builtin_prefetch(inp_b + (size_t)(t+1)*BB*II + tid*(BB*II/128), 0, 0);

    // ---- wait for all z_{t-1} publications ----
    if (tid == 0 && t > 0) {
      while (__hip_atomic_load(&cnt[t-1], __ATOMIC_ACQUIRE,
                               __HIP_MEMORY_SCOPE_AGENT) < NWG)
        __builtin_amdgcn_s_sleep(1);
    }
    __syncthreads();

    // ---- stage z_{t-1} into LDS ----
    {
      const char* g = (const char*)(zslots + (size_t)((t + 3) & 3)*BH); // 64KB
#pragma unroll
      for (int j = 0; j < 32; ++j) {           // one 2048B row per iteration
        const int p = j*128 + tid;
        async_g2l_b128(ZST_OFF + (unsigned)(j*ZPAD*2 + tid*16),
                       g + (size_t)p*16u);
      }
    }
    wait_async0();
    __syncthreads();   // staged frames visible to all waves

    // ---- GEMM part 1: acc += input[t] @ iw^T   (K = 512, all LDS) ----
#pragma unroll 4
    for (int k0 = 0; k0 < II; k0 += 32) {
      union { v16bf v; uint4 q[2]; } a0, a1, bt;
      const __bf16* ap = ist + arow*IPAD + (k0 + kofs);
      a0.q[0] = *(const uint4*)(ap);
      a0.q[1] = *(const uint4*)(ap + 16);
      a1.q[0] = *(const uint4*)(ap + 16*IPAD);
      a1.q[1] = *(const uint4*)(ap + 16*IPAD + 16);
      const __bf16* bp = &ldsw[(k0 + kofs)*WPAD + wave*16 + ln];
#pragma unroll
      for (int s = 0; s < 16; ++s) bt.v[s] = bp[((s & 7) + (s >> 3)*16)*WPAD];
      WMMA_B16(acc0, a0.v, bt.v);
      WMMA_B16(acc1, a1.v, bt.v);
    }
    // ---- GEMM part 2: acc += z_{t-1} @ rw^T   (K = 1024, all LDS) ----
#pragma unroll 4
    for (int k0 = 0; k0 < HH; k0 += 32) {
      union { v16bf v; uint4 q[2]; } a0, a1, bt;
      const __bf16* ap = zst + arow*ZPAD + (k0 + kofs);
      a0.q[0] = *(const uint4*)(ap);
      a0.q[1] = *(const uint4*)(ap + 16);
      a1.q[0] = *(const uint4*)(ap + 16*ZPAD);
      a1.q[1] = *(const uint4*)(ap + 16*ZPAD + 16);
      const __bf16* bp = &ldsw[(II + k0 + kofs)*WPAD + wave*16 + ln];
#pragma unroll
      for (int s = 0; s < 16; ++s) bt.v[s] = bp[((s & 7) + (s >> 3)*16)*WPAD];
      WMMA_B16(acc0, a0.v, bt.v);
      WMMA_B16(acc1, a1.v, bt.v);
    }
#pragma unroll
    for (int e = 0; e < 16; ++e)
      i_s[e] = (e < 8) ? acc0[e] : acc1[e - 8];   // i_new = i_dec + x + z@Wr^T
  }

  // ---- final carry states: (zf, vf, if, bf) ----
  const size_t OFF = (size_t)T_STEPS*BH;
#pragma unroll
  for (int e = 0; e < 16; ++e) {
    const int bi = (e >> 3)*16 + (e & 7) + 8*l16;
    const size_t a = (size_t)bi*HH + hn;
    out[OFF + a]        = z_s[e];
    out[OFF + BH + a]   = v_s[e];
    out[OFF + 2*BH + a] = i_s[e];
    out[OFF + 3*BH + a] = b_s[e];
  }
}

// ---- host launch -----------------------------------------------------------

extern "C" void kernel_launch(void* const* d_in, const int* in_sizes, int n_in,
                              void* d_out, int out_size, void* d_ws, size_t ws_size,
                              hipStream_t stream) {
  (void)in_sizes; (void)n_in; (void)out_size; (void)ws_size;
  const float* inp = (const float*)d_in[0];
  const float* z0  = (const float*)d_in[1];
  const float* v0  = (const float*)d_in[2];
  const float* i0  = (const float*)d_in[3];
  const float* b0  = (const float*)d_in[4];
  const float* iw  = (const float*)d_in[5];
  const float* rw  = (const float*)d_in[6];

  char* ws = (char*)d_ws;
  unsigned* cnt = (unsigned*)(ws + OFF_CNT);
  __bf16* zsl   = (__bf16*)(ws + OFF_Z);
  __bf16* iwT   = (__bf16*)(ws + OFF_IWT);
  __bf16* rwT   = (__bf16*)(ws + OFF_RWT);
  __bf16* inb   = (__bf16*)(ws + OFF_INP);
  float* out    = (float*)d_out;

  k_init<<<(BH + 255)/256, 256, 0, stream>>>(cnt, z0, zsl);
  k_tw  <<<1024, 256, 0, stream>>>(iw, rw, iwT, rwT);
  k_cvt <<<2048, 256, 0, stream>>>(inp, inb, T_STEPS*BB*II);
  lsnn_scan<<<NWG, 128, LDS_BYTES, stream>>>(inb, iwT, rwT, v0, i0, b0,
                                             zsl, cnt, out);
}
